// attention_jpb_32547262169741
// MI455X (gfx1250) — matrix operand order repaired and checked by reading, not compiled, not run on hardware
//
#include <hip/hip_runtime.h>
#include <hip/hip_bf16.h>

// ---------------------------------------------------------------------------
// CDNA5 (gfx1250) implementation of the skeleton-attention block.
// All contractions run on v_wmma_f32_16x16x32_f16 (f16 in, f32 accumulate).
// wave32; each 16x16 output tile is owned by one wave.
// ---------------------------------------------------------------------------

typedef __attribute__((ext_vector_type(16))) _Float16 v16h;
typedef __attribute__((ext_vector_type(8)))  _Float16 v8h;
typedef __attribute__((ext_vector_type(8)))  float    v8f;

#define N_   64
#define C_   96
#define T_   128
#define V_   25
#define S_   3
#define MID  32
#define O3   288                 // 3*C
#define TV   3200                // T*V
#define CTV  (C_*TV)             // 307200
#define OTV  (O3*TV)             // 921600

// joint -> part / body group index tables (precomputed from the reference)
__constant__ int c_jp[25] = {0,0,1,1,2,2,2,3,4,4,4,5,6,6,7,7,8,8,9,9,0,3,3,5,5};
__constant__ int c_jb[25] = {4,4,4,4,0,0,0,0,1,1,1,1,2,2,2,2,3,3,3,3,4,0,0,1,1};

// ---------------------------------------------------------------------------
// WMMA fragment loaders from LDS (row = 32 contiguous f16).
// A (16xK=32, 16-bit): lanes 0-15 hold M=0..15 with K 0-7 / 16-23,
//                      lanes 16-31 hold M=0..15 with K 8-15 / 24-31.
// B (K=32x16, 16-bit): lanes 0-15 hold N=0..15 with K 0-7 / 16-23,
//                      lanes 16-31 hold N=0..15 with K 8-15 / 24-31.
// Both load patterns are two 16B ds reads from an LDS row.
// ---------------------------------------------------------------------------
__device__ inline v16h load_frag_a(const _Float16* row, int lane) {
  const int kb = (lane < 16) ? 0 : 8;
  v8h lo = *(const v8h*)(row + kb);
  v8h hi = *(const v8h*)(row + kb + 16);
  v16h r;
#pragma unroll
  for (int i = 0; i < 8; ++i) { r[i] = lo[i]; r[i + 8] = hi[i]; }
  return r;
}

__device__ inline v16h load_frag_b(const _Float16* row, int lane) {
  const int kb = (lane < 16) ? 0 : 8;
  v8h lo = *(const v8h*)(row + kb);
  v8h hi = *(const v8h*)(row + kb + 16);
  v16h r;
#pragma unroll
  for (int i = 0; i < 8; ++i) { r[i] = lo[i]; r[i + 8] = hi[i]; }
  return r;
}

__device__ inline v8f wmma_f16(v16h a, v16h b, v8f c) {
  return __builtin_amdgcn_wmma_f32_16x16x32_f16(false, a, false, b,
                                                (short)0, c, false, false);
}

// ---------------------------------------------------------------------------
// Prep: f32 weights -> f16, fold BN into per-channel scale/shift.
// ---------------------------------------------------------------------------
__global__ void prep_kernel(const float* __restrict__ w_in,
                            const float* __restrict__ w_ff,
                            const float* __restrict__ gamma,
                            const float* __restrict__ beta,
                            const float* __restrict__ mean,
                            const float* __restrict__ var,
                            _Float16* __restrict__ w_in_h,
                            _Float16* __restrict__ w_ff_h,
                            float* __restrict__ bn_scale,
                            float* __restrict__ bn_shift) {
  int i = blockIdx.x * blockDim.x + threadIdx.x;
  if (i < O3 * C_) w_in_h[i] = (_Float16)w_in[i];
  if (i < C_ * C_) w_ff_h[i] = (_Float16)w_ff[i];
  if (i < C_) {
    float inv   = gamma[i] * rsqrtf(var[i] + 1e-5f);
    bn_scale[i] = inv;
    bn_shift[i] = beta[i] - mean[i] * inv;
  }
}

// ---------------------------------------------------------------------------
// QKV projection: qkv[n,o,tv] = sum_c w_in[o,c] * x[n,c,tv] + b_in[o]
// Block = 32(M) x 64(N) output tile, 8 waves, 1 WMMA tile / wave / k-step.
// grid = (50 ntv-blocks, 9 m-blocks, 64 n)
// ---------------------------------------------------------------------------
__global__ __launch_bounds__(256)
void qkv_kernel(const float* __restrict__ x, const _Float16* __restrict__ w_h,
                const float* __restrict__ b_in, _Float16* __restrict__ qkv) {
  const int nb   = blockIdx.x;
  const int mb   = blockIdx.y;
  const int n    = blockIdx.z;
  const int tid  = threadIdx.x;
  const int lane = tid & 31, wave = tid >> 5;
  const int mt   = wave >> 2, nt = wave & 3;

  __shared__ __align__(16) _Float16 sW[32][32];
  __shared__ __align__(16) _Float16 sX[64][32];

  const float* xn = x + (size_t)n * CTV;
  v8f acc = {};

  for (int kk = 0; kk < C_; kk += 32) {
#pragma unroll
    for (int i = 0; i < 4; ++i) {                 // 1024 halves of W
      int idx = tid + i * 256;
      int m = idx >> 5, k = idx & 31;
      sW[m][k] = w_h[(mb * 32 + m) * C_ + kk + k];
    }
#pragma unroll
    for (int i = 0; i < 8; ++i) {                 // 2048 halves of X (transpose)
      int idx = tid + i * 256;
      int k = idx >> 6, col = idx & 63;
      sX[col][k] = (_Float16)xn[(size_t)(kk + k) * TV + nb * 64 + col];
    }
    __syncthreads();
    v16h a = load_frag_a(&sW[mt * 16 + (lane & 15)][0], lane);
    v16h b = load_frag_b(&sX[nt * 16 + (lane & 15)][0], lane);
    acc = wmma_f16(a, b, acc);
    __syncthreads();
  }

  const int col = nb * 64 + nt * 16 + (lane & 15);
#pragma unroll
  for (int g = 0; g < 8; ++g) {
    int o = mb * 32 + mt * 16 + g + ((lane >= 16) ? 8 : 0);
    float vv = acc[g] + b_in[o];
    qkv[(size_t)n * OTV + (size_t)o * TV + col] = (_Float16)vv;
  }
}

// ---------------------------------------------------------------------------
// Gram partials: part[chunk, ns, u, v] = sum_{c, t in chunk} q[c,t,u]*k[c,t,v]
// K = 4096 split into 8 t-chunks of 16 for parallelism (1536 blocks).
// 128 threads = 4 waves -> 2x2 tiles covering the padded 32x32 output.
// ---------------------------------------------------------------------------
__global__ __launch_bounds__(128)
void gram_kernel(const _Float16* __restrict__ qkv, float* __restrict__ part) {
  const int chunk = blockIdx.x & 7;
  const int ns    = blockIdx.x >> 3;
  const int s = ns % S_, n = ns / S_;
  const int tid = threadIdx.x, lane = tid & 31, wave = tid >> 5;
  const int mt = wave >> 1, nt = wave & 1;

  __shared__ __align__(16) _Float16 sQ[32][32];
  __shared__ __align__(16) _Float16 sK[32][32];

  const _Float16* qb = qkv + (size_t)n * OTV + (size_t)(s * MID) * TV;
  const _Float16* kb = qb + (size_t)C_ * TV;

  v8f acc = {};
  for (int tt = 0; tt < 16; ++tt) {
    const int t = chunk * 16 + tt;
#pragma unroll
    for (int i = 0; i < 8; ++i) {                 // stage 2x 1024 halves
      int idx = tid + i * 128;
      int c = idx >> 5, u = idx & 31;
      sQ[u][c] = (u < V_) ? qb[(size_t)c * TV + t * V_ + u] : (_Float16)0.0f;
      sK[u][c] = (u < V_) ? kb[(size_t)c * TV + t * V_ + u] : (_Float16)0.0f;
    }
    __syncthreads();
    v16h a = load_frag_a(&sQ[mt * 16 + (lane & 15)][0], lane);
    v16h b = load_frag_b(&sK[nt * 16 + (lane & 15)][0], lane);
    acc = wmma_f16(a, b, acc);
    __syncthreads();
  }

  const int v = nt * 16 + (lane & 15);
  float* pb = part + ((size_t)chunk * (N_ * S_) + ns) * 1024;
#pragma unroll
  for (int g = 0; g < 8; ++g) {
    int u = mt * 16 + g + ((lane >= 16) ? 8 : 0);
    pb[u * 32 + v] = acc[g];
  }
}

// ---------------------------------------------------------------------------
// Finalize attention (deterministic 8-way partial sum):
//   att = tanh(sum/4096) + p_att[jp[u],jp[v]] + b_att[jb[u],jb[v]]
// Stored padded 32x32 in f16 (pads = 0), already B^T layout for att x val.
// ---------------------------------------------------------------------------
__global__ __launch_bounds__(256)
void att_finalize_kernel(const float* __restrict__ part,
                         const float* __restrict__ p_att,
                         const float* __restrict__ b_att,
                         _Float16* __restrict__ att) {
  const int ns = blockIdx.x;
#pragma unroll
  for (int i = 0; i < 4; ++i) {
    int idx = threadIdx.x + i * 256;
    int u = idx >> 5, v = idx & 31;
    float sum = 0.f;
#pragma unroll
    for (int c = 0; c < 8; ++c)
      sum += part[((size_t)c * (N_ * S_) + ns) * 1024 + idx];
    float a = 0.f;
    if (u < V_ && v < V_) {
      a = tanhf(sum * (1.0f / 4096.0f));
      a += p_att[(ns * 10 + c_jp[u]) * 10 + c_jp[v]];
      a += b_att[(ns * 5 + c_jb[u]) * 5 + c_jb[v]];
    }
    att[(size_t)ns * 1024 + idx] = (_Float16)a;
  }
}

// ---------------------------------------------------------------------------
// att x val: y2[n, s*32+c, t, u] = sum_v att[u,v] * val[c,t,v]
// Single K=32 step; block = 64 ct-rows x 32 u-cols, 8 waves.
// grid.x = 64*S_*N_ * ... -> (64 ct-blocks) * (N_*S_)
// ---------------------------------------------------------------------------
__global__ __launch_bounds__(256)
void av_kernel(const _Float16* __restrict__ qkv, const _Float16* __restrict__ att,
               _Float16* __restrict__ y2) {
  const int ctb = blockIdx.x & 63;
  const int ns  = blockIdx.x >> 6;
  const int s = ns % S_, n = ns / S_;
  const int tid = threadIdx.x, lane = tid & 31, wave = tid >> 5;
  const int ct_t = wave >> 1, u_t = wave & 1;

  __shared__ __align__(16) _Float16 sV[64][32];
  __shared__ __align__(16) _Float16 sA[32][32];

  const _Float16* vb = qkv + (size_t)n * OTV + (size_t)(2 * C_ + s * MID) * TV;
  const _Float16* ab = att + (size_t)ns * 1024;
#pragma unroll
  for (int i = 0; i < 4; ++i)                      // att tile, contiguous copy
    (&sA[0][0])[tid + i * 256] = ab[tid + i * 256];
#pragma unroll
  for (int i = 0; i < 8; ++i) {                    // val tile (zero-pad v>=25)
    int idx = tid + i * 256;
    int r = idx >> 5, v = idx & 31;
    int ct = ctb * 64 + r;
    int c = ct >> 7, t = ct & 127;
    sV[r][v] = (v < V_) ? vb[(size_t)c * TV + t * V_ + v] : (_Float16)0.0f;
  }
  __syncthreads();

  v16h a = load_frag_a(&sV[ct_t * 16 + (lane & 15)][0], lane);
  v16h b = load_frag_b(&sA[u_t * 16 + (lane & 15)][0], lane);
  v8f acc = {};
  acc = wmma_f16(a, b, acc);

  const int u = u_t * 16 + (lane & 15);
  if (u < V_) {
#pragma unroll
    for (int g = 0; g < 8; ++g) {
      int r = ct_t * 16 + g + ((lane >= 16) ? 8 : 0);
      int ct = ctb * 64 + r;
      int c = ct >> 7, t = ct & 127;
      y2[(size_t)n * CTV + (size_t)(s * MID + c) * TV + t * V_ + u] =
          (_Float16)acc[g];
    }
  }
}

// ---------------------------------------------------------------------------
// FF projection + bias + BN + residual + leaky_relu(0.1), f32 out.
// Same GEMM shape as qkv (96x96 x 96x3200 per n); grid = (50, 3, 64).
// ---------------------------------------------------------------------------
__global__ __launch_bounds__(256)
void ff_kernel(const _Float16* __restrict__ y2, const _Float16* __restrict__ w_h,
               const float* __restrict__ b_ff, const float* __restrict__ bn_scale,
               const float* __restrict__ bn_shift, const float* __restrict__ x,
               float* __restrict__ out) {
  const int nb   = blockIdx.x;
  const int mb   = blockIdx.y;
  const int n    = blockIdx.z;
  const int tid  = threadIdx.x;
  const int lane = tid & 31, wave = tid >> 5;
  const int mt   = wave >> 2, nt = wave & 3;

  __shared__ __align__(16) _Float16 sW[32][32];
  __shared__ __align__(16) _Float16 sY[64][32];

  const _Float16* yn = y2 + (size_t)n * CTV;
  v8f acc = {};

  for (int kk = 0; kk < C_; kk += 32) {
#pragma unroll
    for (int i = 0; i < 4; ++i) {
      int idx = tid + i * 256;
      int m = idx >> 5, k = idx & 31;
      sW[m][k] = w_h[(mb * 32 + m) * C_ + kk + k];
    }
#pragma unroll
    for (int i = 0; i < 8; ++i) {
      int idx = tid + i * 256;
      int k = idx >> 6, col = idx & 63;
      sY[col][k] = yn[(size_t)(kk + k) * TV + nb * 64 + col];
    }
    __syncthreads();
    v16h a = load_frag_a(&sW[mt * 16 + (lane & 15)][0], lane);
    v16h b = load_frag_b(&sY[nt * 16 + (lane & 15)][0], lane);
    acc = wmma_f16(a, b, acc);
    __syncthreads();
  }

  const int col = nb * 64 + nt * 16 + (lane & 15);
#pragma unroll
  for (int g = 0; g < 8; ++g) {
    int o = mb * 32 + mt * 16 + g + ((lane >= 16) ? 8 : 0);
    float vv = acc[g] + b_ff[o];
    vv = vv * bn_scale[o] + bn_shift[o];
    float z = x[(size_t)n * CTV + (size_t)o * TV + col] + vv;
    out[(size_t)n * CTV + (size_t)o * TV + col] = (z >= 0.f) ? z : 0.1f * z;
  }
}

// ---------------------------------------------------------------------------
// Workspace layout (bytes), ~157 MB total:
//   0          : w_in_h  (55,296)
//   65536      : w_ff_h  (18,432)
//   98304      : bn_scale (384) ; 98816 : bn_shift (384)
//   131072     : qkv f16  (117,964,800)
//   118095872  : att f16  (393,216)
//   118489088  : y2  f16  (39,321,600)
//   157810688  : gram partials f32 (6,291,456)
// ---------------------------------------------------------------------------
extern "C" void kernel_launch(void* const* d_in, const int* in_sizes, int n_in,
                              void* d_out, int out_size, void* d_ws, size_t ws_size,
                              hipStream_t stream) {
  (void)in_sizes; (void)n_in; (void)out_size; (void)ws_size;
  const float* x     = (const float*)d_in[0];
  const float* p_att = (const float*)d_in[1];
  const float* b_att = (const float*)d_in[2];
  const float* w_in  = (const float*)d_in[3];
  const float* b_in  = (const float*)d_in[4];
  const float* w_ff  = (const float*)d_in[5];
  const float* b_ff  = (const float*)d_in[6];
  const float* gamma = (const float*)d_in[7];
  const float* beta  = (const float*)d_in[8];
  const float* mean  = (const float*)d_in[9];
  const float* var   = (const float*)d_in[10];

  char* ws = (char*)d_ws;
  _Float16* w_in_h  = (_Float16*)(ws + 0);
  _Float16* w_ff_h  = (_Float16*)(ws + 65536);
  float*    bnscale = (float*)(ws + 98304);
  float*    bnshift = (float*)(ws + 98816);
  _Float16* qkv     = (_Float16*)(ws + 131072);
  _Float16* att     = (_Float16*)(ws + 118095872);
  _Float16* y2      = (_Float16*)(ws + 118489088);
  float*    part    = (float*)(ws + 157810688);

  prep_kernel<<<dim3(108), dim3(256), 0, stream>>>(
      w_in, w_ff, gamma, beta, mean, var, w_in_h, w_ff_h, bnscale, bnshift);

  qkv_kernel<<<dim3(50, 9, 64), dim3(256), 0, stream>>>(x, w_in_h, b_in, qkv);

  gram_kernel<<<dim3(N_ * S_ * 8), dim3(128), 0, stream>>>(qkv, part);

  att_finalize_kernel<<<dim3(N_ * S_), dim3(256), 0, stream>>>(
      part, p_att, b_att, att);

  av_kernel<<<dim3(64 * N_ * S_), dim3(256), 0, stream>>>(qkv, att, y2);

  ff_kernel<<<dim3(50, 3, 64), dim3(256), 0, stream>>>(
      y2, w_ff_h, b_ff, bnscale, bnshift, x, (float*)d_out);
}
